// DynamicGCNLayer_17961553231914
// MI455X (gfx1250) — matrix operand order repaired and checked by reading, not compiled, not run on hardware
//
#include <hip/hip_runtime.h>
#include <stdint.h>

// ---- problem constants (fixed by the reference) ----
#define B_ 8
#define N_ 512
#define D_ 128   // IN_DIM == HID == OUT_DIM == 128

typedef __attribute__((ext_vector_type(16))) __bf16 v16bf;
typedef __attribute__((ext_vector_type(8)))  float  v8f;

union V16U {
    uint32_t u[8];
    v16bf    v;
};

// round-to-nearest-even float -> bf16 bits (finite inputs) — used once for w1
__device__ __forceinline__ uint32_t f2bf_bits(float f) {
    union { float f; uint32_t u; } c;
    c.f = f;
    uint32_t u = c.u;
    u += 0x7fffu + ((u >> 16) & 1u);
    return u >> 16;
}

// pack {|d1|,|d0|} as two bf16 in one dword:
//   v_perm_b32 grabs the two high halves, one v_and clears both signs.
// (truncating bf16 conversion; <=1 ulp vs RNE, 4 VALU per pair total)
__device__ __forceinline__ uint32_t pack_absbf(float d0, float d1) {
    union { float f; uint32_t u; } a, b;
    a.f = d0;
    b.f = d1;
    // result = { b[31:24], b[23:16], a[31:24], a[23:16] }
    uint32_t p = __builtin_amdgcn_perm(b.u, a.u, 0x07060302u);
    return p & 0x7fff7fffu;
}

// -------------------------------------------------------------------------
// Kernel 0: zero degree accumulator + convert w1 -> bf16 bits, transposed
// [h][d].  grid 64 x 256 threads = 16384 = 128*128 elements
// -------------------------------------------------------------------------
__global__ void prep_kernel(const float* __restrict__ w1,
                            unsigned short* __restrict__ w1T,
                            float* __restrict__ deg) {
    int idx = blockIdx.x * 256 + threadIdx.x;     // 0..16383
    if (idx < B_ * N_) deg[idx] = 0.0f;
    int d = idx >> 7;          // input dim
    int h = idx & 127;         // hidden dim
    w1T[h * 128 + d] = (unsigned short)f2bf_bits(w1[idx]);  // w1 stored [d][h]
}

// -------------------------------------------------------------------------
// Kernel 1: xw[b,n,o] = sum_k x[b,n,k] * gcn_w[k,o]   (fp32, tiny GEMM)
// grid = B*N/2 blocks of 256 threads (2 rows per block)
// -------------------------------------------------------------------------
__global__ void xw_kernel(const float* __restrict__ x,
                          const float* __restrict__ gcn_w,
                          float* __restrict__ xw) {
    __shared__ float sx[2][128];
    const int half = threadIdx.x >> 7;
    const int o    = threadIdx.x & 127;
    const int row  = blockIdx.x * 2 + half;       // flattened (b,n)
    sx[half][o] = x[row * 128 + o];
    __syncthreads();
    float acc = 0.0f;
#pragma unroll 8
    for (int k = 0; k < 128; ++k)
        acc += sx[half][k] * gcn_w[k * 128 + o];
    xw[row * 128 + o] = acc;
}

// -------------------------------------------------------------------------
// Kernel 2: fused edge-MLP via WMMA bf16.
//   Each wave owns edges (i, j0..j0+15):  scores = w2 . relu(w1^T|x_i-x_j|+b1)
//   adj = sigmoid(scores); deg row-sums accumulated atomically.
// Block = 256 threads (8 waves -> i = i_base..i_base+7), grid (N/16, N/8, B)
// -------------------------------------------------------------------------
__global__ void __launch_bounds__(256)
edge_kernel(const float* __restrict__ x,
            const unsigned short* __restrict__ w1T,   // [h][d] bf16 bits
            const float* __restrict__ b1,
            const float* __restrict__ w2,
            const float* __restrict__ b2,
            float* __restrict__ adj,
            float* __restrict__ deg) {
    __shared__ unsigned short sW[128 * 128];      // 32 KB  w1T staged
    __shared__ float  sXj[16 * 128];              //  8 KB  x_j tile
    __shared__ float  sXi[8 * 128];               //  4 KB  x_i rows (1/wave)

    const int tid = threadIdx.x;
    const int b  = blockIdx.z;
    const int i_base = blockIdx.y * 8;
    const int j0     = blockIdx.x * 16;

    // cooperative staging
    const uint32_t* w1Tu = (const uint32_t*)w1T;
    uint32_t* sWu = (uint32_t*)sW;
#pragma unroll
    for (int t = 0; t < 32; ++t)                  // 8192 dwords
        sWu[t * 256 + tid] = w1Tu[t * 256 + tid];
#pragma unroll
    for (int t = 0; t < 8; ++t) {                 // 2048 floats
        int idx = t * 256 + tid;
        sXj[idx] = x[((size_t)b * N_ + j0 + (idx >> 7)) * 128 + (idx & 127)];
    }
#pragma unroll
    for (int t = 0; t < 4; ++t) {                 // 1024 floats
        int idx = t * 256 + tid;
        sXi[idx] = x[((size_t)b * N_ + i_base + (idx >> 7)) * 128 + (idx & 127)];
    }
    __syncthreads();

    const int wv   = tid >> 5;
    const int lane = tid & 31;
    const int i    = i_base + wv;
    const int m    = lane & 15;                   // A-matrix M row (= j offset)
    const int g    = lane >> 4;                   // lane-half
    const float* xi  = &sXi[wv * 128];
    const float* xjm = &sXj[m * 128];

    v8f acc[8];
    v8f zero = {0.f, 0.f, 0.f, 0.f, 0.f, 0.f, 0.f, 0.f};
#pragma unroll
    for (int t = 0; t < 8; ++t) acc[t] = zero;

#pragma unroll
    for (int ks = 0; ks < 4; ++ks) {              // K = 128 in steps of 32
        const int kbase = ks * 32;
        // A fragment: contiguous k order (same as B below).
        //   lane m, elem e: K = kbase + e + 16*g
        // pairs (2t, 2t+1) -> consecutive k, packed via v_perm + sign-clear.
        V16U af;
#pragma unroll
        for (int t = 0; t < 8; ++t) {
            const int e0 = 2 * t;
            const int k0 = kbase + e0 + 16 * g;
            af.u[t] = pack_absbf(xi[k0] - xjm[k0], xi[k0 + 1] - xjm[k0 + 1]);
        }
#pragma unroll
        for (int nt = 0; nt < 8; ++nt) {          // hidden dim in tiles of 16
            const int col = nt * 16 + m;          // B-matrix N column
            // B fragment: lane holds K = kbase+16g .. +15 contiguous
            const uint32_t* wp = (const uint32_t*)&sW[col * 128 + kbase + 16 * g];
            V16U bf;
#pragma unroll
            for (int t = 0; t < 8; ++t) bf.u[t] = wp[t];
            acc[nt] = __builtin_amdgcn_wmma_f32_16x16x32_bf16(
                false, af.v, false, bf.v, (short)0, acc[nt], false, false);
        }
    }

    // Epilogue: h = acc + b1 ; relu ; dot w2  -> per-edge score
    // C layout: VGPR r, lane l: (M = r + 8*(l>=16), N = l&15)
    float s[8];
#pragma unroll
    for (int r = 0; r < 8; ++r) s[r] = 0.0f;
#pragma unroll
    for (int nt = 0; nt < 8; ++nt) {
        const int n = nt * 16 + m;
        const float w2n = w2[n];
        const float b1n = b1[n];
#pragma unroll
        for (int r = 0; r < 8; ++r) {
            float h = acc[nt][r] + b1n;
            h = fmaxf(h, 0.0f);
            s[r] += h * w2n;
        }
    }
    // reduce over N within each 16-lane half
#pragma unroll
    for (int r = 0; r < 8; ++r)
        for (int msk = 1; msk < 16; msk <<= 1)
            s[r] += __shfl_xor(s[r], msk, 32);

    const float b2v = b2[0];
    float contrib = 0.0f;
    if (m < 8) {                                   // 16 storing lanes / wave
        const int r = m;
        const float sc = s[r] + b2v;
        const float av = 1.0f / (1.0f + __expf(-sc));
        const int j = j0 + r + 8 * g;              // edge M = r + 8g
        adj[((size_t)(b * N_ + i)) * N_ + j] = av;
        contrib = av;
    }
    // wave-wide row-sum for degree
    for (int msk = 1; msk < 32; msk <<= 1)
        contrib += __shfl_xor(contrib, msk, 32);
    if (lane == 0)
        atomicAdd(&deg[b * N_ + i], contrib);
}

// -------------------------------------------------------------------------
// Kernel 3: out[b,i,o] = dinv_i * sum_j adj[b,i,j]*dinv_j*xw[b,j,o] + gcn_b[o]
// grid (N, B), 128 threads (one per output channel)
// -------------------------------------------------------------------------
__global__ void gcn_kernel(const float* __restrict__ adj,
                           const float* __restrict__ deg,
                           const float* __restrict__ xw,
                           const float* __restrict__ gcn_b,
                           float* __restrict__ out) {
    __shared__ float sDinv[N_];
    __shared__ float sAdj[N_];
    const int i = blockIdx.x;
    const int b = blockIdx.y;
    const int o = threadIdx.x;                    // 0..127
    for (int idx = o; idx < N_; idx += 128) {
        sDinv[idx] = rsqrtf(deg[b * N_ + idx]);
        sAdj[idx]  = adj[((size_t)(b * N_ + i)) * N_ + idx];
    }
    __syncthreads();
    float acc = 0.0f;
#pragma unroll 4
    for (int j = 0; j < N_; ++j)
        acc += sAdj[j] * sDinv[j] * xw[((size_t)b * N_ + j) * 128 + o];
    out[((size_t)b * N_ + i) * 128 + o] = sDinv[i] * acc + gcn_b[o];
}

// -------------------------------------------------------------------------
extern "C" void kernel_launch(void* const* d_in, const int* in_sizes, int n_in,
                              void* d_out, int out_size, void* d_ws, size_t ws_size,
                              hipStream_t stream) {
    const float* x     = (const float*)d_in[0];
    const float* w1    = (const float*)d_in[1];
    const float* b1    = (const float*)d_in[2];
    const float* w2    = (const float*)d_in[3];
    const float* b2    = (const float*)d_in[4];
    const float* gcn_w = (const float*)d_in[5];
    const float* gcn_b = (const float*)d_in[6];

    float* out = (float*)d_out;                        // [B,N,128]
    float* adj = out + (size_t)B_ * N_ * 128;          // [B,N,N]

    // workspace: deg (16KB) | xw (2MB) | w1T bf16 (32KB)  -> ~2.1MB total
    float*          deg = (float*)d_ws;
    float*          xw  = deg + B_ * N_;
    unsigned short* w1T = (unsigned short*)(xw + (size_t)B_ * N_ * 128);

    prep_kernel<<<64, 256, 0, stream>>>(w1, w1T, deg);
    xw_kernel<<<B_ * N_ / 2, 256, 0, stream>>>(x, gcn_w, xw);

    dim3 eg(N_ / 16, N_ / 8, B_);                      // (32, 64, 8)
    edge_kernel<<<eg, 256, 0, stream>>>(x, w1T, b1, w2, b2, adj, deg);

    dim3 gg(N_, B_);
    gcn_kernel<<<gg, 128, 0, stream>>>(adj, deg, xw, gcn_b, out);
}
